// TransformerDecoderLayer_9637906612508
// MI455X (gfx1250) — compile-verified, matrix operand order checked by reading, not run on hardware
//
#include <hip/hip_runtime.h>
#include <cstdint>

// ---------------- problem constants ----------------
#define BSZ 4
#define NQ 256
#define HWN 4096
#define DMODEL 256
#define NH 8
#define DH 32
#define DFF 2048
#define LBIN 24
#define TWOL 48
#define AXL 144            // 3*48
#define GRID_SZ 0.04f
#define EPSV 1e-5f
#define QSCALE 0.17677669529663687f   // DH^-0.5

typedef __attribute__((ext_vector_type(16))) _Float16     v16h;
typedef __attribute__((ext_vector_type(8)))  float        v8f;
typedef __attribute__((ext_vector_type(4)))  unsigned int v4u;
typedef __attribute__((ext_vector_type(4)))  int          v4i;
typedef __attribute__((ext_vector_type(8)))  int          v8i;

__device__ __forceinline__ v8f wmma16(v16h a, v16h b, v8f c) {
  // D = A(16x32,f16) * B(32x16,f16) + C(16x16,f32)
  return __builtin_amdgcn_wmma_f32_16x16x32_f16(false, a, false, b, (short)0, c, false, false);
}

__device__ __forceinline__ int bin_of(float rel) {
  int bi = (int)floorf(rel * (1.0f / GRID_SZ));
  bi = bi < -LBIN ? -LBIN : bi;
  bi = bi > (LBIN - 1) ? (LBIN - 1) : bi;
  return bi + LBIN;
}

// ---------------- TDM: DMA a 2D f32 tile (tile_w x tile_h, row stride in elems)
// from global memory into LDS at lds_byte_addr. D# layout per CDNA5 ISA 8.3/8.4.
__device__ __forceinline__ void tdm_load_2d_f32(
    unsigned int lds_byte_addr, const void* gptr,
    unsigned int tile_w, unsigned int tile_h, unsigned int row_stride_elems)
{
  uint64_t ga = (uint64_t)(uintptr_t)gptr;
  v4u g0;
  g0[0] = 1u;                                  // count=1, user descriptor, no gather
  g0[1] = lds_byte_addr;                       // D#.lds_addr
  g0[2] = (unsigned int)ga;                    // D#.global_addr[31:0]
  g0[3] = (unsigned int)(ga >> 32) | (2u << 30); // addr[56:32] | type=2 ("image")
  v8i g1;
  g1[0] = (int)(2u << 16);                     // workgroup_mask=0, data_size=2 (4B)
  g1[1] = (int)((tile_w & 0xffffu) << 16);     // tensor_dim0[15:0]  (== tile_w)
  g1[2] = (int)((tile_w >> 16) | ((tile_h & 0xffffu) << 16));   // td0 hi | tensor_dim1 lo
  g1[3] = (int)((tile_h >> 16) | (tile_w << 16));               // td1 hi | tile_dim0
  g1[4] = (int)(tile_h & 0xffffu);             // tile_dim1 (tile_dim2 = 0)
  g1[5] = (int)row_stride_elems;               // tensor_dim0_stride lo32
  g1[6] = 0;                                   // stride hi16 | dim1_stride lo16
  g1[7] = 0;
  v4i z4 = {0, 0, 0, 0};
#if __has_include(<hip/amd_detail/amd_gfx1250_TDM.h>)
  v8i z8 = {0, 0, 0, 0, 0, 0, 0, 0};
  __builtin_amdgcn_tensor_load_to_lds(g0, g1, z4, z4, z8, 0);   // therock 6-arg form
#else
  __builtin_amdgcn_tensor_load_to_lds(g0, g1, z4, z4, 0);       // ROCm 7.2 5-arg form
#endif
}

__device__ __forceinline__ unsigned int lds_addr_of(const void* p) {
  // LDS aperture: hardware maps flat addr[31:0] onto the LDS allocation.
  return (unsigned int)(uintptr_t)p;
}

// A-fragment element e (lane half hf) covers K = base + hf*8 + (e<8?0:16) + e%8
// B-fragment uses the same K-map with lane%16 selecting the column.

// ---------------- generic WMMA GEMM: C = act((A@W + bias [+ add]) * scale) ------
// A: row-major MxK (f32), W: row-major KxN (f32). Block 256 thr = 8 waves = 64x32
// tile. W k-slabs are double-buffered through LDS by the TDM: wave 0 issues the
// DMA for slab i+1, waits tensorcnt<=1 (slab i complete, in-order), barrier
// publishes, all 8 waves feed WMMA B-fragments from LDS while slab i+1 flies.
__global__ __launch_bounds__(256) void gemm_wmma_kernel(
    const float* __restrict__ A, const float* __restrict__ W,
    const float* __restrict__ bias, const float* __restrict__ addsrc,
    float* __restrict__ C, int M, int K, int N, float scale, int relu)
{
  __shared__ float s_w[2][32 * 32];   // double-buffered k-local(32) x n-local(32)
  int lane = threadIdx.x & 31;
  int wave = threadIdx.x >> 5;
  int wm = wave & 3, wn = wave >> 2;
  int tileM = blockIdx.x * 64 + wm * 16;
  int tileN = blockIdx.y * 32 + wn * 16;
  int hf  = lane >> 4;
  int l15 = lane & 15;
  const float* Arow = A + (size_t)(tileM + l15) * K;  // this lane's A row
  int nloc = wn * 16 + l15;                           // B column within 32-wide slab
  const float* Wtile = W + blockIdx.y * 32;
  int nsteps = K / 32;
  if (wave == 0)
    tdm_load_2d_f32(lds_addr_of(&s_w[0][0]), Wtile, 32, 32, (unsigned)N);
  v8f acc = {};
  for (int i = 0; i < nsteps; ++i) {
    if (wave == 0) {
      if (i + 1 < nsteps) {   // prefetch next slab, then wait only for slab i
        tdm_load_2d_f32(lds_addr_of(&s_w[(i + 1) & 1][0]),
                        Wtile + (size_t)(i + 1) * 32 * N, 32, 32, (unsigned)N);
        __builtin_amdgcn_s_wait_tensorcnt(1);
      } else {
        __builtin_amdgcn_s_wait_tensorcnt(0);
      }
    }
    __syncthreads();                        // slab i visible to all waves
    const float* sw = &s_w[i & 1][0];
    int k0 = i * 32 + hf * 8;
    __builtin_prefetch(Arow + k0 + 64, 0, 1);
    v16h af, bf;
#pragma unroll
    for (int e = 0; e < 8; ++e) {
      af[e]     = (_Float16)Arow[k0 + e];
      af[e + 8] = (_Float16)Arow[k0 + 16 + e];
    }
#pragma unroll
    for (int e = 0; e < 8; ++e) {
      bf[e]     = (_Float16)sw[(hf * 8 + e) * 32 + nloc];
      bf[e + 8] = (_Float16)sw[(hf * 8 + 16 + e) * 32 + nloc];
    }
    acc = wmma16(af, bf, acc);
    __syncthreads();                        // slab i consumed; safe to overwrite later
  }
#pragma unroll
  for (int r = 0; r < 8; ++r) {
    int row = tileM + r + 8 * hf;
    int col = tileN + l15;
    float v = acc[r] + bias[col];
    if (addsrc) v += addsrc[(size_t)row * N + col];
    v *= scale;
    if (relu) v = fmaxf(v, 0.f);
    C[(size_t)row * N + col] = v;
  }
}

// ---------------- (B,S,D) -> (S,B,D) transpose ----------------
__global__ void swap01_kernel(const float* __restrict__ X, float* __restrict__ Y,
                              int B, int S)
{
  size_t idx = (size_t)blockIdx.x * blockDim.x + threadIdx.x;
  size_t total = (size_t)B * S * DMODEL;
  if (idx >= total) return;
  int d = (int)(idx % DMODEL);
  size_t r = idx / DMODEL;
  int s = (int)(r % S);
  int b = (int)(r / S);
  Y[((size_t)s * B + b) * DMODEL + d] = X[idx];
}

// ---------------- self attention (per b,h; 2 waves x 16 q-rows) ----------------
// Q/K/V in "flat" (seq*BS, D) layout; head h occupies cols [h*DH, h*DH+32).
__global__ __launch_bounds__(64) void self_attn_kernel(
    const float* __restrict__ Qf, const float* __restrict__ Kf,
    const float* __restrict__ Vf, float* __restrict__ Ctx)
{
  __shared__ float    s_sc[2][16][NQ];      // 32 KB scores
  __shared__ _Float16 s_at[2][16][NQ];      // 16 KB attn(f16)
  int lane = threadIdx.x & 31, wave = threadIdx.x >> 5;
  int b = blockIdx.x / NH, h = blockIdx.x % NH;
  int qbase = blockIdx.y * 32 + wave * 16;
  int hf = lane >> 4, l15 = lane & 15;

  const float* qrow = Qf + ((size_t)(qbase + l15) * BSZ + b) * DMODEL + h * DH;
  v16h aq;
  int k0 = hf * 8;
#pragma unroll
  for (int e = 0; e < 8; ++e) { aq[e] = (_Float16)qrow[k0 + e]; aq[e + 8] = (_Float16)qrow[k0 + 16 + e]; }

  for (int kt = 0; kt < NQ / 16; ++kt) {
    const float* krow = Kf + ((size_t)(kt * 16 + l15) * BSZ + b) * DMODEL + h * DH;
    v16h bk;
#pragma unroll
    for (int e = 0; e < 8; ++e) { bk[e] = (_Float16)krow[k0 + e]; bk[e + 8] = (_Float16)krow[k0 + 16 + e]; }
    v8f c = {};
    c = wmma16(aq, bk, c);
#pragma unroll
    for (int r = 0; r < 8; ++r) s_sc[wave][r + 8 * hf][kt * 16 + l15] = c[r];
  }
  __syncthreads();
  if (lane < 16) {               // one lane per query row
    float mx = -1e30f;
    for (int j = 0; j < NQ; ++j) mx = fmaxf(mx, s_sc[wave][lane][j]);
    float sum = 0.f;
    for (int j = 0; j < NQ; ++j) { float e = __expf(s_sc[wave][lane][j] - mx); s_sc[wave][lane][j] = e; sum += e; }
    float inv = 1.f / sum;
    for (int j = 0; j < NQ; ++j) s_at[wave][lane][j] = (_Float16)(s_sc[wave][lane][j] * inv);
  }
  __syncthreads();
  v8f c0 = {}, c1 = {};
  for (int kc = 0; kc < NQ / 32; ++kc) {
    int kk = kc * 32 + hf * 8;
    v16h aa, b0, b1;
#pragma unroll
    for (int e = 0; e < 8; ++e) { aa[e] = s_at[wave][l15][kk + e]; aa[e + 8] = s_at[wave][l15][kk + 16 + e]; }
#pragma unroll
    for (int e = 0; e < 8; ++e) {
      const float* v0 = Vf + ((size_t)(kk + e) * BSZ + b) * DMODEL + h * DH;
      const float* v1 = Vf + ((size_t)(kk + 16 + e) * BSZ + b) * DMODEL + h * DH;
      b0[e] = (_Float16)v0[l15];      b0[e + 8] = (_Float16)v1[l15];
      b1[e] = (_Float16)v0[16 + l15]; b1[e + 8] = (_Float16)v1[16 + l15];
    }
    c0 = wmma16(aa, b0, c0);
    c1 = wmma16(aa, b1, c1);
  }
#pragma unroll
  for (int r = 0; r < 8; ++r) {
    int i = qbase + r + 8 * hf;
    float* dst = Ctx + ((size_t)i * BSZ + b) * DMODEL + h * DH;  // unheads layout
    dst[l15] = c0[r];
    dst[16 + l15] = c1[r];
  }
}

// ---------------- qT / kT : (b,h,s,ax*48+l) = sum_d x_h[d] * T[h][d][axl] ------
__global__ void proj_T_kernel(const float* __restrict__ Xf, const float* __restrict__ T,
                              float* __restrict__ Out, int S)
{
  size_t idx = (size_t)blockIdx.x * blockDim.x + threadIdx.x;
  size_t total = (size_t)BSZ * NH * S * AXL;
  if (idx >= total) return;
  int axl = (int)(idx % AXL);
  size_t r = idx / AXL;
  int s = (int)(r % S);
  size_t bh = r / S;
  int h = (int)(bh % NH), b = (int)(bh / NH);
  const float* xr = Xf + ((size_t)s * BSZ + b) * DMODEL + h * DH;
  const float* tr = T + (size_t)h * DH * AXL + axl;   // stride AXL over d
  float acc = 0.f;
#pragma unroll 8
  for (int d = 0; d < DH; ++d) acc += xr[d] * tr[(size_t)d * AXL];
  Out[idx] = acc;
}

// ---------------- cross-attn scores: qk + binned qT/kT biases ----------------
__global__ __launch_bounds__(256) void ca_scores_kernel(
    const float* __restrict__ Q2, const float* __restrict__ K2,
    const float* __restrict__ qT, const float* __restrict__ kT,
    const float* __restrict__ qcf /*(NQ,BS,3)*/, const float* __restrict__ kcf /*(BS,HW,3)*/,
    float* __restrict__ Scores)
{
  __shared__ float s_qc[16][3];
  __shared__ float s_qT[16][AXL];
  int bh = blockIdx.z;
  int b = bh / NH;
  int itile = blockIdx.y * 16;
  int jbase = blockIdx.x * 128;
  for (int t = threadIdx.x; t < 16 * AXL; t += 256) {
    int il = t / AXL, axl = t % AXL;
    s_qT[il][axl] = qT[((size_t)bh * NQ + itile + il) * AXL + axl];
  }
  if (threadIdx.x < 48) {
    int il = threadIdx.x / 3, ax = threadIdx.x % 3;
    s_qc[il][ax] = qcf[((size_t)(itile + il) * BSZ + b) * 3 + ax];
  }
  __syncthreads();

  int lane = threadIdx.x & 31, wave = threadIdx.x >> 5;
  int hf = lane >> 4, l15 = lane & 15;
  int jt = jbase + wave * 16;
  int k0 = hf * 8;
  const float* qrow = Q2 + ((size_t)(itile + l15) * BSZ + b) * DMODEL + (bh % NH) * DH;
  const float* krow = K2 + ((size_t)(jt + l15) * BSZ + b) * DMODEL + (bh % NH) * DH;
  v16h aq, bk;
#pragma unroll
  for (int e = 0; e < 8; ++e) { aq[e] = (_Float16)qrow[k0 + e]; aq[e + 8] = (_Float16)qrow[k0 + 16 + e]; }
#pragma unroll
  for (int e = 0; e < 8; ++e) { bk[e] = (_Float16)krow[k0 + e]; bk[e + 8] = (_Float16)krow[k0 + 16 + e]; }
  v8f c = {};
  c = wmma16(aq, bk, c);

  int j = jt + l15;
  float kc0 = kcf[((size_t)b * HWN + j) * 3 + 0];
  float kc1 = kcf[((size_t)b * HWN + j) * 3 + 1];
  float kc2 = kcf[((size_t)b * HWN + j) * 3 + 2];
  const float* kTrow = kT + ((size_t)bh * HWN + j) * AXL;
#pragma unroll
  for (int r = 0; r < 8; ++r) {
    int il = r + 8 * hf;
    float v = c[r];
    int b0 = bin_of(s_qc[il][0] - kc0);
    int b1 = bin_of(s_qc[il][1] - kc1);
    int b2 = bin_of(s_qc[il][2] - kc2);
    v += s_qT[il][b0]            + kTrow[b0];
    v += s_qT[il][TWOL + b1]     + kTrow[TWOL + b1];
    v += s_qT[il][2 * TWOL + b2] + kTrow[2 * TWOL + b2];
    Scores[((size_t)bh * NQ + itile + il) * HWN + j] = v;
  }
}

// ---------------- row softmax over HW=4096 (in place) ----------------
__global__ __launch_bounds__(256) void softmax_kernel(float* __restrict__ S)
{
  __shared__ float red[256];
  float* row = S + (size_t)blockIdx.x * HWN;
  float vals[HWN / 256];
  float mx = -1e30f;
#pragma unroll
  for (int t = 0; t < HWN / 256; ++t) { vals[t] = row[threadIdx.x + t * 256]; mx = fmaxf(mx, vals[t]); }
  red[threadIdx.x] = mx; __syncthreads();
  for (int s = 128; s > 0; s >>= 1) { if (threadIdx.x < s) red[threadIdx.x] = fmaxf(red[threadIdx.x], red[threadIdx.x + s]); __syncthreads(); }
  mx = red[0]; __syncthreads();
  float sum = 0.f;
#pragma unroll
  for (int t = 0; t < HWN / 256; ++t) { vals[t] = __expf(vals[t] - mx); sum += vals[t]; }
  red[threadIdx.x] = sum; __syncthreads();
  for (int s = 128; s > 0; s >>= 1) { if (threadIdx.x < s) red[threadIdx.x] += red[threadIdx.x + s]; __syncthreads(); }
  float inv = 1.f / red[0];
#pragma unroll
  for (int t = 0; t < HWN / 256; ++t) row[threadIdx.x + t * 256] = vals[t] * inv;
}

// ---------------- src_weight = mean over heads ----------------
__global__ void srcw_kernel(const float* __restrict__ Attn, float* __restrict__ Out)
{
  size_t idx = (size_t)blockIdx.x * blockDim.x + threadIdx.x;
  if (idx >= (size_t)BSZ * NQ * HWN) return;
  size_t j = idx % HWN;
  size_t bi = idx / HWN;
  size_t i = bi % NQ, b = bi / NQ;
  float s = 0.f;
#pragma unroll
  for (int h = 0; h < NH; ++h) s += Attn[((b * NH + h) * NQ + i) * HWN + j];
  Out[idx] = s * (1.f / NH);
}

// ---------------- A[b,h,i,ax,l] = scatter-add of attn over bins (LDS atomics) --
__global__ __launch_bounds__(256) void a_scatter_kernel(
    const float* __restrict__ Attn, const float* __restrict__ qcf,
    const float* __restrict__ kcf, float* __restrict__ Aout)
{
  __shared__ float bins[AXL];
  int bhi = blockIdx.x;
  int i = bhi % NQ;
  int bh = bhi / NQ;
  int b = bh / NH;
  if (threadIdx.x < AXL) bins[threadIdx.x] = 0.f;
  __syncthreads();
  float qc0 = qcf[((size_t)i * BSZ + b) * 3 + 0];
  float qc1 = qcf[((size_t)i * BSZ + b) * 3 + 1];
  float qc2 = qcf[((size_t)i * BSZ + b) * 3 + 2];
  const float* arow = Attn + (size_t)bhi * HWN;
  for (int j = threadIdx.x; j < HWN; j += 256) {
    float a = arow[j];
    const float* kc = kcf + ((size_t)b * HWN + j) * 3;
    atomicAdd(&bins[bin_of(qc0 - kc[0])], a);
    atomicAdd(&bins[TWOL + bin_of(qc1 - kc[1])], a);
    atomicAdd(&bins[2 * TWOL + bin_of(qc2 - kc[2])], a);
  }
  __syncthreads();
  if (threadIdx.x < AXL) Aout[(size_t)bhi * AXL + threadIdx.x] = bins[threadIdx.x];
}

// ---------------- ctx = attn @ v_heads (WMMA over K=HW) ----------------
// V slabs (32 j-rows x 32 d-cols, row stride BS*D) are double-buffered through
// LDS by the TDM with the same issue-ahead / tensorcnt<=1 pipeline as the GEMM.
__global__ __launch_bounds__(64) void ca_ctx_kernel(
    const float* __restrict__ Attn, const float* __restrict__ V2, float* __restrict__ Ctx)
{
  __shared__ float s_v[2][32 * 32];
  int lane = threadIdx.x & 31, wave = threadIdx.x >> 5;
  int bh = blockIdx.x;
  int b = bh / NH, h = bh % NH;
  int qbase = blockIdx.y * 32 + wave * 16;
  int hf = lane >> 4, l15 = lane & 15;
  const float* arow = Attn + ((size_t)bh * NQ + qbase + l15) * HWN;
  const float* Vbase = V2 + (size_t)b * DMODEL + h * DH;   // + j*BS*D for row j
  int nsteps = HWN / 32;
  if (wave == 0)
    tdm_load_2d_f32(lds_addr_of(&s_v[0][0]), Vbase, 32, 32, BSZ * DMODEL);
  v8f c0 = {}, c1 = {};
  for (int i = 0; i < nsteps; ++i) {
    if (wave == 0) {
      if (i + 1 < nsteps) {
        tdm_load_2d_f32(lds_addr_of(&s_v[(i + 1) & 1][0]),
                        Vbase + (size_t)(i + 1) * 32 * BSZ * DMODEL, 32, 32, BSZ * DMODEL);
        __builtin_amdgcn_s_wait_tensorcnt(1);
      } else {
        __builtin_amdgcn_s_wait_tensorcnt(0);
      }
    }
    __syncthreads();                 // slab i visible
    const float* sv = &s_v[i & 1][0];
    int kk = i * 32 + hf * 8;
    __builtin_prefetch(arow + kk + 128, 0, 1);
    v16h aa, b0, b1;
#pragma unroll
    for (int e = 0; e < 8; ++e) { aa[e] = (_Float16)arow[kk + e]; aa[e + 8] = (_Float16)arow[kk + 16 + e]; }
#pragma unroll
    for (int e = 0; e < 8; ++e) {
      int j0 = hf * 8 + e, j1 = hf * 8 + 16 + e;   // j-local rows of slab
      b0[e] = (_Float16)sv[j0 * 32 + l15];      b0[e + 8] = (_Float16)sv[j1 * 32 + l15];
      b1[e] = (_Float16)sv[j0 * 32 + 16 + l15]; b1[e + 8] = (_Float16)sv[j1 * 32 + 16 + l15];
    }
    c0 = wmma16(aa, b0, c0);
    c1 = wmma16(aa, b1, c1);
    __syncthreads();                 // slab i consumed
  }
#pragma unroll
  for (int r = 0; r < 8; ++r) {
    int i = qbase + r + 8 * hf;
    float* dst = Ctx + ((size_t)i * BSZ + b) * DMODEL + h * DH;   // unheads layout
    dst[l15] = c0[r];
    dst[16 + l15] = c1[r];
  }
}

// ---------------- ctx += A . Tv ----------------
__global__ __launch_bounds__(32) void ctx_tv_kernel(
    const float* __restrict__ A, const float* __restrict__ Tv, float* __restrict__ Ctx)
{
  int bhi = blockIdx.x;
  int i = bhi % NQ;
  int bh = bhi / NQ;
  int b = bh / NH, h = bh % NH;
  int d = threadIdx.x;
  const float* arow = A + (size_t)bhi * AXL;
  const float* trow = Tv + ((size_t)h * DH + d) * AXL;
  float s = 0.f;
#pragma unroll 8
  for (int t = 0; t < AXL; ++t) s += arow[t] * trow[t];
  Ctx[((size_t)i * BSZ + b) * DMODEL + h * DH + d] += s;
}

// ---------------- y = LN(x + r) * g + bn ; optional (S,B,D)->(B,S,D) out ------
__global__ __launch_bounds__(256) void add_ln_kernel(
    const float* __restrict__ X, const float* __restrict__ R,
    const float* __restrict__ g, const float* __restrict__ bn,
    float* __restrict__ Y, int transpose_out)
{
  __shared__ float red[256];
  int row = blockIdx.x;
  int d = threadIdx.x;
  float v = X[(size_t)row * DMODEL + d] + R[(size_t)row * DMODEL + d];
  red[d] = v; __syncthreads();
  for (int s = 128; s > 0; s >>= 1) { if (d < s) red[d] += red[d + s]; __syncthreads(); }
  float mean = red[0] * (1.f / DMODEL); __syncthreads();
  float c = v - mean;
  red[d] = c * c; __syncthreads();
  for (int s = 128; s > 0; s >>= 1) { if (d < s) red[d] += red[d + s]; __syncthreads(); }
  float var = red[0] * (1.f / DMODEL);
  float y = c * rsqrtf(var + EPSV) * g[d] + bn[d];
  if (transpose_out) {
    int b = row % BSZ, i = row / BSZ;
    Y[((size_t)b * NQ + i) * DMODEL + d] = y;
  } else {
    Y[(size_t)row * DMODEL + d] = y;
  }
}

// ================== host side ==================
#define WPAR(i) ((const float*)d_in[6 + 2 * (i)])
#define BPAR(i) ((const float*)d_in[7 + 2 * (i)])

static inline void gemm(const float* A, const float* W, const float* b, const float* add,
                        float* C, int M, int K, int N, float scale, int relu, hipStream_t s)
{
  dim3 g(M / 64, N / 32);
  gemm_wmma_kernel<<<g, dim3(256), 0, s>>>(A, W, b, add, C, M, K, N, scale, relu);
}

extern "C" void kernel_launch(void* const* d_in, const int* in_sizes, int n_in,
                              void* d_out, int out_size, void* d_ws, size_t ws_size,
                              hipStream_t stream)
{
  (void)in_sizes; (void)n_in; (void)out_size; (void)ws_size;
  const float* tgt   = (const float*)d_in[0];   // (BS,NQ,D)
  const float* mem   = (const float*)d_in[1];   // (BS,HW,D)
  const float* qcf   = (const float*)d_in[2];   // (NQ,BS,3)
  const float* kcf   = (const float*)d_in[3];   // (BS,HW,3)
  const float* pos   = (const float*)d_in[4];   // (HW,BS,D) - already flat rows j*BS+b
  const float* qpos  = (const float*)d_in[5];   // (NQ,BS,D)
  const float* Tq    = (const float*)d_in[38];
  const float* Tk    = (const float*)d_in[39];
  const float* Tv    = (const float*)d_in[40];
  const float* W1    = (const float*)d_in[41];
  const float* b1    = (const float*)d_in[42];
  const float* W2    = (const float*)d_in[43];
  const float* b2    = (const float*)d_in[44];
  float* out_t  = (float*)d_out;                             // (BS,NQ,D)
  float* out_sw = (float*)d_out + (size_t)BSZ * NQ * DMODEL; // (BS,NQ,HW)

  const size_t SMALL = (size_t)NQ * BSZ * DMODEL;   // 262144
  const size_t BIG   = (size_t)HWN * BSZ * DMODEL;  // 4194304
  char* p = (char*)d_ws;
  auto alloc = [&](size_t n) { float* r = (float*)p; p += n * sizeof(float); return r; };
  float* t0      = alloc(SMALL);
  float* memf    = alloc(BIG);
  float* t1      = alloc(SMALL);
  float* t2      = alloc(SMALL);
  float* tmp1    = alloc(SMALL);
  float* qbuf    = alloc(SMALL);
  float* sa_q    = alloc(SMALL);
  float* sa_k    = alloc(SMALL);
  float* sa_v    = alloc(SMALL);
  float* sa_ctx  = alloc(SMALL);
  float* q2      = alloc(SMALL);
  float* ktmp    = alloc(BIG);
  float* kbuf    = alloc(BIG);
  float* k2      = alloc(BIG);
  float* vbuf    = alloc(BIG);
  float* v2      = alloc(BIG);
  float* qTw     = alloc((size_t)BSZ * NH * NQ * AXL);
  float* kTw     = alloc((size_t)BSZ * NH * HWN * AXL);
  float* scores  = alloc((size_t)BSZ * NH * NQ * HWN);
  float* Aw      = alloc((size_t)BSZ * NH * NQ * AXL);
  float* cactx   = alloc(SMALL);
  float* ffh     = alloc((size_t)NQ * BSZ * DFF);
  float* ff      = alloc(SMALL);

  // ---- layout: (B,S,D) -> (S,B,D) flats ----
  swap01_kernel<<<(unsigned)((SMALL + 255) / 256), 256, 0, stream>>>(tgt, t0, BSZ, NQ);
  swap01_kernel<<<(unsigned)((BIG + 255) / 256), 256, 0, stream>>>(mem, memf, BSZ, HWN);

  // ---- self attention ----
  gemm(t0, WPAR(0), BPAR(0), nullptr, tmp1, 1024, 256, 256, 1.f, 0, stream);       // sa_qc
  gemm(tmp1, WPAR(3), BPAR(3), nullptr, sa_q, 1024, 256, 256, QSCALE, 0, stream);  // sa_qi * scale
  gemm(t0, WPAR(1), BPAR(1), nullptr, tmp1, 1024, 256, 256, 1.f, 0, stream);       // sa_kc
  gemm(tmp1, WPAR(4), BPAR(4), nullptr, sa_k, 1024, 256, 256, 1.f, 0, stream);     // sa_ki
  gemm(t0, WPAR(2), BPAR(2), nullptr, tmp1, 1024, 256, 256, 1.f, 0, stream);       // sa_v
  gemm(tmp1, WPAR(5), BPAR(5), nullptr, sa_v, 1024, 256, 256, 1.f, 0, stream);     // sa_vi
  self_attn_kernel<<<dim3(BSZ * NH, NQ / 32), 64, 0, stream>>>(sa_q, sa_k, sa_v, sa_ctx);
  gemm(sa_ctx, WPAR(6), BPAR(6), nullptr, tmp1, 1024, 256, 256, 1.f, 0, stream);   // sa_o
  add_ln_kernel<<<1024, 256, 0, stream>>>(t0, tmp1, (const float*)d_in[45], (const float*)d_in[46], t1, 0);

  // ---- cross attention projections ----
  gemm(t1, WPAR(7), BPAR(7), nullptr, tmp1, 1024, 256, 256, 1.f, 0, stream);       // ca_qc
  gemm(qpos, WPAR(8), BPAR(8), tmp1, qbuf, 1024, 256, 256, 1.f, 0, stream);        // + ca_qp
  gemm(qbuf, WPAR(12), BPAR(12), nullptr, q2, 1024, 256, 256, QSCALE, 0, stream);  // ca_qi * scale
  gemm(memf, WPAR(9), BPAR(9), nullptr, ktmp, 16384, 256, 256, 1.f, 0, stream);    // ca_kc
  gemm(pos, WPAR(10), BPAR(10), ktmp, kbuf, 16384, 256, 256, 1.f, 0, stream);      // + ca_kp
  gemm(kbuf, WPAR(13), BPAR(13), nullptr, k2, 16384, 256, 256, 1.f, 0, stream);    // ca_ki
  gemm(memf, WPAR(11), BPAR(11), nullptr, vbuf, 16384, 256, 256, 1.f, 0, stream);  // ca_v
  gemm(vbuf, WPAR(14), BPAR(14), nullptr, v2, 16384, 256, 256, 1.f, 0, stream);    // ca_vi

  // ---- bin-bias tables ----
  proj_T_kernel<<<(unsigned)(((size_t)BSZ * NH * NQ * AXL) / 256), 256, 0, stream>>>(q2, Tq, qTw, NQ);
  proj_T_kernel<<<(unsigned)(((size_t)BSZ * NH * HWN * AXL) / 256), 256, 0, stream>>>(k2, Tk, kTw, HWN);

  // ---- scores + softmax + attn-derived outputs ----
  ca_scores_kernel<<<dim3(HWN / 128, NQ / 16, BSZ * NH), 256, 0, stream>>>(q2, k2, qTw, kTw, qcf, kcf, scores);
  softmax_kernel<<<BSZ * NH * NQ, 256, 0, stream>>>(scores);
  srcw_kernel<<<(unsigned)(((size_t)BSZ * NQ * HWN) / 256), 256, 0, stream>>>(scores, out_sw);
  a_scatter_kernel<<<BSZ * NH * NQ, 256, 0, stream>>>(scores, qcf, kcf, Aw);
  ca_ctx_kernel<<<dim3(BSZ * NH, NQ / 32), 64, 0, stream>>>(scores, v2, cactx);
  ctx_tv_kernel<<<BSZ * NH * NQ, 32, 0, stream>>>(Aw, Tv, cactx);

  // ---- output projection + LN2 ----
  gemm(cactx, WPAR(15), BPAR(15), nullptr, tmp1, 1024, 256, 256, 1.f, 0, stream);  // ca_o
  add_ln_kernel<<<1024, 256, 0, stream>>>(t1, tmp1, (const float*)d_in[47], (const float*)d_in[48], t2, 0);

  // ---- FFN + LN3 (write transposed straight to d_out) ----
  gemm(t2, W1, b1, nullptr, ffh, 1024, 256, 2048, 1.f, 1, stream);
  gemm(ffh, W2, b2, nullptr, ff, 1024, 2048, 256, 1.f, 0, stream);
  add_ln_kernel<<<1024, 256, 0, stream>>>(t2, ff, (const float*)d_in[49], (const float*)d_in[50], out_t, 1);
}
